// PointCloudEncoderGeomRadar_86268713107574
// MI455X (gfx1250) — matrix operand order repaired and checked by reading, not compiled, not run on hardware
//
#include <hip/hip_runtime.h>
#include <hip/hip_bf16.h>
#include <math.h>

typedef __attribute__((ext_vector_type(16))) __bf16 v16bf;
typedef __attribute__((ext_vector_type(8)))  float  v8f;

#define NB      8
#define NPTS    4096
#define P_TOTAL 32768          // B*N flattened point axis
#define KNN     8
#define EPSBN   1e-5f

#define GF_RELU  1
#define GF_RESID 2

// ---------------------------------------------------------------------------
// Pointwise-conv GEMM: Y[o,p] = sum_c W[o,c]*X[c,p] + bias[o]
// bf16 WMMA 16x16x32, fp32 accumulate.
// Block = 256 threads = 8 waves as 4(o) x 2(p) grid of 16x16 tiles
// -> block tile 64(o) x 32(p). W/X K-slabs staged in LDS via coalesced
// loads (float4 for X, scalar for W); fragments built from LDS (conflict-free ds reads).
// CIN is compile-time so the K-loop has no bounds checks.
// ---------------------------------------------------------------------------
template <int CIN>
__global__ __launch_bounds__(256) void pw_gemm_wmma(
    const float* __restrict__ W, const float* __restrict__ X,
    const float* __restrict__ bias, float* __restrict__ Y, int flags)
{
  __shared__ float Ws[64 * 32];          // 8 KB
  __shared__ float Xs[32 * 32];          // 4 KB
  const int tid  = threadIdx.x;
  const int wv   = tid >> 5, lane = tid & 31;
  const int l    = lane & 15, half = lane >> 4;
  const int ow   = wv >> 1;              // 0..3  (o sub-tile)
  const int pw_i = wv & 1;               // 0..1  (p sub-tile)
  const int blocksP = P_TOTAL / 32;
  const int to = (blockIdx.x / blocksP) * 64;
  const int p0 = (blockIdx.x % blocksP) * 32;

  v8f acc;
  #pragma unroll
  for (int r = 0; r < 8; ++r) {
    const int o = to + ow * 16 + r + 8 * half;
    float c0 = bias ? bias[o] : 0.0f;
    if (flags & GF_RESID) c0 += Y[(size_t)o * P_TOTAL + p0 + pw_i * 16 + l];
    acc[r] = c0;
  }

  for (int k0 = 0; k0 < CIN; k0 += 32) {
    // stage W tile 64 x 32 (rows of 32 contiguous -> coalesced)
    #pragma unroll
    for (int i = 0; i < 8; ++i) {
      const int idx = i * 256 + tid;
      const int r = idx >> 5, c = idx & 31;
      Ws[idx] = W[(size_t)(to + r) * CIN + k0 + c];
    }
    // stage X slab 32 x 32 with one float4 per thread (coalesced)
    {
      const float4* X4 = (const float4*)(X + (size_t)k0 * P_TOTAL + p0);
      float4* Xs4 = (float4*)Xs;
      const int row = tid >> 3, c4 = tid & 7;
      Xs4[tid] = X4[(size_t)row * (P_TOTAL / 4) + c4];
    }
    __syncthreads();

    v16bf a, b;
    #pragma unroll
    for (int j = 0; j < 8; ++j) {
      const int kb = ((j < 4) ? 0 : 16) + 8 * half + 2 * (j & 3);
      a[2 * j]     = (__bf16)Ws[(ow * 16 + l) * 32 + kb];
      a[2 * j + 1] = (__bf16)Ws[(ow * 16 + l) * 32 + kb + 1];
      const int kr = ((j < 4) ? 0 : 16) + 8 * half + 2 * (j & 3);
      b[2 * j]     = (__bf16)Xs[kr * 32 + pw_i * 16 + l];
      b[2 * j + 1] = (__bf16)Xs[(kr + 1) * 32 + pw_i * 16 + l];
    }
    acc = __builtin_amdgcn_wmma_f32_16x16x32_bf16(false, a, false, b,
                                                  (short)0, acc, false, false);
    __syncthreads();
  }

  #pragma unroll
  for (int r = 0; r < 8; ++r) {
    const int o = to + ow * 16 + r + 8 * half;
    float v = acc[r];
    if (flags & GF_RELU) v = fmaxf(v, 0.0f);
    Y[(size_t)o * P_TOTAL + p0 + pw_i * 16 + l] = v;
  }
}

// ---------------------------------------------------------------------------
// BatchNorm statistics / apply.
// ---------------------------------------------------------------------------
__global__ __launch_bounds__(256) void bn_stats_kernel(
    const float* __restrict__ Y, float* __restrict__ mv)
{
  const int c = blockIdx.x;
  __shared__ float ssum[256], ssq[256];
  float s = 0.0f, q = 0.0f;
  for (int p = threadIdx.x; p < P_TOTAL; p += 256) {
    const float v = Y[(size_t)c * P_TOTAL + p];
    s += v; q += v * v;
  }
  ssum[threadIdx.x] = s; ssq[threadIdx.x] = q;
  __syncthreads();
  for (int st = 128; st > 0; st >>= 1) {
    if (threadIdx.x < st) {
      ssum[threadIdx.x] += ssum[threadIdx.x + st];
      ssq[threadIdx.x]  += ssq[threadIdx.x + st];
    }
    __syncthreads();
  }
  if (threadIdx.x == 0) {
    const float m = ssum[0] * (1.0f / P_TOTAL);
    const float v = ssq[0] * (1.0f / P_TOTAL) - m * m;
    mv[2 * c] = m; mv[2 * c + 1] = v;
  }
}

__global__ void bn_apply_relu_kernel(float* __restrict__ Y,
                                     const float* __restrict__ mv,
                                     const float* __restrict__ g,
                                     const float* __restrict__ be, int Cout)
{
  const size_t i = (size_t)blockIdx.x * blockDim.x + threadIdx.x;
  if (i >= (size_t)Cout * P_TOTAL) return;
  const int c = (int)(i / P_TOTAL);
  const float m = mv[2 * c], v = mv[2 * c + 1];
  const float y = (Y[i] - m) * rsqrtf(v + EPSBN) * g[c] + be[c];
  Y[i] = fmaxf(y, 0.0f);
}

__global__ void zero_fill_kernel(float* __restrict__ p, size_t n)
{
  const size_t i = (size_t)blockIdx.x * blockDim.x + threadIdx.x;
  if (i < n) p[i] = 0.0f;
}

// ---------------------------------------------------------------------------
// kNN geometric feature. One wave per query; coords staged in LDS (48 KB);
// per-lane sorted top-8 then 32-lane tournament merge (wave32 shuffles).
// ---------------------------------------------------------------------------
__global__ __launch_bounds__(256) void knn_xc_kernel(
    const float* __restrict__ x, float* __restrict__ xc)
{
  __shared__ float cs[NPTS * 3];
  const int b  = blockIdx.x / (NPTS / 8);
  const int q0 = (blockIdx.x % (NPTS / 8)) * 8;
  for (int i = threadIdx.x; i < NPTS * 3; i += 256) {
    const int n = i / 3, d = i % 3;
    cs[i] = x[((size_t)b * NPTS + n) * 5 + d];
  }
  __syncthreads();

  const int wave = threadIdx.x >> 5, lane = threadIdx.x & 31;
  const int q = q0 + wave;
  const float qx = cs[q * 3 + 0], qy = cs[q * 3 + 1], qz = cs[q * 3 + 2];

  float bd[KNN]; int bi[KNN];
  #pragma unroll
  for (int j = 0; j < KNN; ++j) { bd[j] = 3.4e38f; bi[j] = 0; }

  for (int m = lane; m < NPTS; m += 32) {
    if (m == q) continue;
    const float dx = cs[m * 3] - qx, dy = cs[m * 3 + 1] - qy, dz = cs[m * 3 + 2] - qz;
    const float d2 = dx * dx + dy * dy + dz * dz;
    if (d2 < bd[KNN - 1]) {
      int j = KNN - 1;
      while (j > 0 && bd[j - 1] > d2) { bd[j] = bd[j - 1]; bi[j] = bi[j - 1]; --j; }
      bd[j] = d2; bi[j] = m;
    }
  }

  int ptr = 0;
  float sx = 0.0f, sy = 0.0f, sz = 0.0f;
  for (int s = 0; s < KNN; ++s) {
    float bv = bd[ptr]; int bl = lane; int bidx = bi[ptr];
    #pragma unroll
    for (int off = 16; off > 0; off >>= 1) {
      const float ov = __shfl_xor(bv, off, 32);
      const int   ol = __shfl_xor(bl, off, 32);
      const int   oi = __shfl_xor(bidx, off, 32);
      if (ov < bv) { bv = ov; bl = ol; bidx = oi; }
    }
    if (lane == bl) ++ptr;
    sx += cs[bidx * 3]; sy += cs[bidx * 3 + 1]; sz += cs[bidx * 3 + 2];
  }
  if (lane == 0) {
    const size_t p = (size_t)b * NPTS + q;
    xc[0 * (size_t)P_TOTAL + p] = sx * (1.0f / KNN) - qx;
    xc[1 * (size_t)P_TOTAL + p] = sy * (1.0f / KNN) - qy;
    xc[2 * (size_t)P_TOTAL + p] = sz * (1.0f / KNN) - qz;
  }
}

__global__ void extract_attrs_kernel(const float* __restrict__ x,
                                     float* __restrict__ xa)
{
  const size_t i = (size_t)blockIdx.x * blockDim.x + threadIdx.x;
  if (i >= (size_t)2 * P_TOTAL) return;
  const int c = (int)(i / P_TOTAL);
  const size_t p = i % P_TOTAL;
  xa[(size_t)c * P_TOTAL + p] = x[p * 5 + 3 + c];
}

// ---------------------------------------------------------------------------
// Flash attention, head dim 128, N=4096 keys/queries per batch.
// Block = 8 waves covering 128 consecutive queries of one batch; all waves
// consume the same K/V tiles, so each 128(h) x 32(m) K and V tile is staged
// once per block into LDS with coalesced float4 loads (2 x 16 KB).
// S^T orientation keeps exp(S) in-lane for the P.V^T A-fragment.
// Q,K,V,O: [128][P_TOTAL] fp32.
// ---------------------------------------------------------------------------
__global__ __launch_bounds__(256) void flash_attn_kernel(
    const float* __restrict__ Q, const float* __restrict__ Km,
    const float* __restrict__ V, float* __restrict__ O)
{
  __shared__ float Ks[128 * 32];   // [h][m] 16 KB
  __shared__ float Vs[128 * 32];   // [h][m] 16 KB
  const int tid  = threadIdx.x;
  const int wv   = tid >> 5, lane = tid & 31;
  const int l    = lane & 15, half = lane >> 4;
  const int blocksN = NPTS / 128;
  const int b  = blockIdx.x / blocksN;
  const int n0 = (blockIdx.x % blocksN) * 128 + wv * 16;
  const size_t colQ = (size_t)b * NPTS + n0;

  // Preload Q as four B-fragments (h contraction, 4 x K=32 steps).
  v16bf qb[4];
  #pragma unroll
  for (int ks = 0; ks < 4; ++ks)
    #pragma unroll
    for (int j = 0; j < 8; ++j)
      #pragma unroll
      for (int t = 0; t < 2; ++t) {
        const int h = ks * 32 + 16 * half + 2 * j + t;
        qb[ks][2 * j + t] = (__bf16)Q[(size_t)h * P_TOTAL + colQ + l];
      }

  v8f o_acc[8];
  #pragma unroll
  for (int ht = 0; ht < 8; ++ht) o_acc[ht] = (v8f){};
  float mrow = -3.4e38f;
  float lrow = 0.0f;

  for (int m0 = 0; m0 < NPTS; m0 += 32) {
    const size_t colK = (size_t)b * NPTS + m0;
    // stage K,V tiles: 4096 floats each -> 4 float4 per thread, coalesced
    {
      const float4* K4 = (const float4*)(Km);
      const float4* V4 = (const float4*)(V);
      float4* Ks4 = (float4*)Ks;
      float4* Vs4 = (float4*)Vs;
      const size_t c4 = colK >> 2;
      #pragma unroll
      for (int i = 0; i < 4; ++i) {
        const int idx4 = i * 256 + tid;          // 0..1023
        const int row = idx4 >> 3, cc = idx4 & 7;
        Ks4[idx4] = K4[(size_t)row * (P_TOTAL / 4) + c4 + cc];
        Vs4[idx4] = V4[(size_t)row * (P_TOTAL / 4) + c4 + cc];
      }
    }
    __syncthreads();

    // S'[m,n] = sum_h K[h,m] Q[h,n] : two 16x16 tiles
    v8f s0 = {}, s1 = {};
    #pragma unroll
    for (int ks = 0; ks < 4; ++ks) {
      v16bf a0, a1;
      #pragma unroll
      for (int j = 0; j < 8; ++j) {
        const int kb = 16 * half + 2 * j;
        #pragma unroll
        for (int t = 0; t < 2; ++t) {
          const int h = ks * 32 + kb + t;
          a0[2 * j + t] = (__bf16)Ks[h * 32 + l];
          a1[2 * j + t] = (__bf16)Ks[h * 32 + 16 + l];
        }
      }
      s0 = __builtin_amdgcn_wmma_f32_16x16x32_bf16(false, a0, false, qb[ks],
                                                   (short)0, s0, false, false);
      s1 = __builtin_amdgcn_wmma_f32_16x16x32_bf16(false, a1, false, qb[ks],
                                                   (short)0, s1, false, false);
    }

    // Online softmax over m, per column n = lane&15 (both halves combined).
    float tmax = -3.4e38f;
    #pragma unroll
    for (int r = 0; r < 8; ++r) tmax = fmaxf(tmax, fmaxf(s0[r], s1[r]));
    tmax = fmaxf(tmax, __shfl_xor(tmax, 16, 32));
    const float mnew = fmaxf(mrow, tmax);
    const float corr = __expf(mrow - mnew);
    float ex0[8], ex1[8], tsum = 0.0f;
    #pragma unroll
    for (int r = 0; r < 8; ++r) {
      ex0[r] = __expf(s0[r] - mnew);
      ex1[r] = __expf(s1[r] - mnew);
      tsum += ex0[r] + ex1[r];
    }
    tsum += __shfl_xor(tsum, 16, 32);
    lrow = lrow * corr + tsum;
    mrow = mnew;

    // P A-fragment (n x 32): entirely in-lane by construction.
    v16bf pa;
    #pragma unroll
    for (int j = 0; j < 4; ++j)
      #pragma unroll
      for (int t = 0; t < 2; ++t) pa[2 * j + t] = (__bf16)ex0[2 * j + t];
    #pragma unroll
    for (int j = 4; j < 8; ++j)
      #pragma unroll
      for (int t = 0; t < 2; ++t) pa[2 * j + t] = (__bf16)ex1[2 * (j - 4) + t];

    // Rescale O rows (n = r + 8*half): factor lives at lane n.
    float corr_r[8];
    #pragma unroll
    for (int r = 0; r < 8; ++r) corr_r[r] = __shfl(corr, 8 * half + r, 32);
    #pragma unroll
    for (int ht = 0; ht < 8; ++ht)
      #pragma unroll
      for (int r = 0; r < 8; ++r) o_acc[ht][r] *= corr_r[r];

    // O[n,h] += P(n x 32) . V^T(32 x h): 8 h-tiles of 16.
    #pragma unroll
    for (int ht = 0; ht < 8; ++ht) {
      v16bf vb;
      #pragma unroll
      for (int j = 0; j < 8; ++j)
        #pragma unroll
        for (int t = 0; t < 2; ++t) {
          const int m = ((j < 4) ? 0 : 16) + 8 * half + 2 * (j & 3) + t;
          vb[2 * j + t] = (__bf16)Vs[(ht * 16 + l) * 32 + m];
        }
      o_acc[ht] = __builtin_amdgcn_wmma_f32_16x16x32_bf16(
          false, pa, false, vb, (short)0, o_acc[ht], false, false);
    }
    __syncthreads();
  }

  float linv_r[8];
  #pragma unroll
  for (int r = 0; r < 8; ++r)
    linv_r[r] = 1.0f / __shfl(lrow, 8 * half + r, 32);
  #pragma unroll
  for (int ht = 0; ht < 8; ++ht)
    #pragma unroll
    for (int r = 0; r < 8; ++r)
      O[(size_t)(ht * 16 + l) * P_TOTAL + colQ + r + 8 * half] =
          o_acc[ht][r] * linv_r[r];
}

// ---------------------------------------------------------------------------
// Pooling tail.
// ---------------------------------------------------------------------------
__global__ void pool_score_kernel(const float* __restrict__ hid,
                                  const float* __restrict__ W2,
                                  const float* __restrict__ b2,
                                  float* __restrict__ w)
{
  const int p = blockIdx.x * blockDim.x + threadIdx.x;
  if (p >= P_TOTAL) return;
  float s = b2[0];
  #pragma unroll 4
  for (int c = 0; c < 64; ++c) s += W2[c] * hid[(size_t)c * P_TOTAL + p];
  w[p] = s;
}

__global__ __launch_bounds__(256) void pool_softmax_kernel(float* __restrict__ w)
{
  const int b = blockIdx.x, tid = threadIdx.x;
  __shared__ float red[256];
  float m = -3.4e38f;
  for (int n = tid; n < NPTS; n += 256) m = fmaxf(m, w[b * NPTS + n]);
  red[tid] = m; __syncthreads();
  for (int st = 128; st > 0; st >>= 1) {
    if (tid < st) red[tid] = fmaxf(red[tid], red[tid + st]);
    __syncthreads();
  }
  m = red[0]; __syncthreads();
  float s = 0.0f;
  for (int n = tid; n < NPTS; n += 256) {
    const float e = __expf(w[b * NPTS + n] - m);
    w[b * NPTS + n] = e; s += e;
  }
  red[tid] = s; __syncthreads();
  for (int st = 128; st > 0; st >>= 1) {
    if (tid < st) red[tid] += red[tid + st];
    __syncthreads();
  }
  const float inv = 1.0f / red[0];
  for (int n = tid; n < NPTS; n += 256) w[b * NPTS + n] *= inv;
}

__global__ __launch_bounds__(256) void pool_reduce_kernel(
    const float* __restrict__ xall, const float* __restrict__ w,
    float* __restrict__ pooled)
{
  const int b = blockIdx.x / 512, c = blockIdx.x % 512;
  __shared__ float red[256];
  float s = 0.0f;
  for (int n = threadIdx.x; n < NPTS; n += 256)
    s += xall[(size_t)c * P_TOTAL + (size_t)b * NPTS + n] * w[b * NPTS + n];
  red[threadIdx.x] = s; __syncthreads();
  for (int st = 128; st > 0; st >>= 1) {
    if (threadIdx.x < st) red[threadIdx.x] += red[threadIdx.x + st];
    __syncthreads();
  }
  if (threadIdx.x == 0) pooled[b * 512 + c] = red[0];
}

__global__ void final_fc_kernel(const float* __restrict__ pooled,
                                const float* __restrict__ Wf,
                                const float* __restrict__ bf,
                                float* __restrict__ out)
{
  const int i = blockIdx.x * blockDim.x + threadIdx.x;
  if (i >= NB * 256) return;
  const int b = i / 256, o = i % 256;
  float s = bf[o];
  #pragma unroll 4
  for (int c = 0; c < 512; ++c) s += pooled[b * 512 + c] * Wf[o * 512 + c];
  out[i] = fmaxf(s, 0.0f);
}

// ---------------------------------------------------------------------------
// Host orchestration.
// ---------------------------------------------------------------------------
static inline void run_gemm(const float* W, const float* X, const float* bias,
                            float* Y, int Cin, int Cout, int flags,
                            hipStream_t stream)
{
  const int blocks = (Cout / 64) * (P_TOTAL / 32);
  switch (Cin) {
    case 32:  pw_gemm_wmma<32> <<<blocks, 256, 0, stream>>>(W, X, bias, Y, flags); break;
    case 64:  pw_gemm_wmma<64> <<<blocks, 256, 0, stream>>>(W, X, bias, Y, flags); break;
    case 128: pw_gemm_wmma<128><<<blocks, 256, 0, stream>>>(W, X, bias, Y, flags); break;
    case 256: pw_gemm_wmma<256><<<blocks, 256, 0, stream>>>(W, X, bias, Y, flags); break;
    default:  pw_gemm_wmma<512><<<blocks, 256, 0, stream>>>(W, X, bias, Y, flags); break;
  }
}

static inline void run_bn(float* Y, float* mv, const float* g, const float* be,
                          int Cout, hipStream_t stream)
{
  bn_stats_kernel<<<Cout, 256, 0, stream>>>(Y, mv);
  const size_t n = (size_t)Cout * P_TOTAL;
  bn_apply_relu_kernel<<<(int)((n + 255) / 256), 256, 0, stream>>>(Y, mv, g, be, Cout);
}

extern "C" void kernel_launch(void* const* d_in, const int* in_sizes, int n_in,
                              void* d_out, int out_size, void* d_ws, size_t ws_size,
                              hipStream_t stream)
{
  (void)in_sizes; (void)n_in; (void)out_size; (void)ws_size;
  const float* x = (const float*)d_in[0];
  const float* cW[3]  = {(const float*)d_in[1],  (const float*)d_in[5],  (const float*)d_in[9]};
  const float* cB[3]  = {(const float*)d_in[2],  (const float*)d_in[6],  (const float*)d_in[10]};
  const float* cG[3]  = {(const float*)d_in[3],  (const float*)d_in[7],  (const float*)d_in[11]};
  const float* cBe[3] = {(const float*)d_in[4],  (const float*)d_in[8],  (const float*)d_in[12]};
  const float* aW[3]  = {(const float*)d_in[13], (const float*)d_in[17], (const float*)d_in[21]};
  const float* aB[3]  = {(const float*)d_in[14], (const float*)d_in[18], (const float*)d_in[22]};
  const float* aG[3]  = {(const float*)d_in[15], (const float*)d_in[19], (const float*)d_in[23]};
  const float* aBe[3] = {(const float*)d_in[16], (const float*)d_in[20], (const float*)d_in[24]};
  const float* Wq = (const float*)d_in[25]; const float* bq = (const float*)d_in[26];
  const float* Wk = (const float*)d_in[27]; const float* bk = (const float*)d_in[28];
  const float* Wv = (const float*)d_in[29]; const float* bv = (const float*)d_in[30];
  const float* Wfc = (const float*)d_in[31]; const float* bfc = (const float*)d_in[32];
  const float* Wp1 = (const float*)d_in[33]; const float* bp1 = (const float*)d_in[34];
  const float* Wp2 = (const float*)d_in[35]; const float* bp2 = (const float*)d_in[36];
  const float* Wf  = (const float*)d_in[37]; const float* bf  = (const float*)d_in[38];

  float* ws = (float*)d_ws;
  size_t off = 0;
  auto alloc = [&](size_t n) { float* p = ws + off; off += n; return p; };
  float* xc     = alloc((size_t)32 * P_TOTAL);    // zero-padded 3 -> 32 channels
  float* xa     = alloc((size_t)32 * P_TOTAL);    // zero-padded 2 -> 32 channels
  float* bufA   = alloc((size_t)128 * P_TOTAL);
  float* bufB   = alloc((size_t)128 * P_TOTAL);
  float* xall   = alloc((size_t)512 * P_TOTAL);   // coord rows 0..255, attr rows 256..511
  float* Qb     = alloc((size_t)128 * P_TOTAL);
  float* Kb     = alloc((size_t)128 * P_TOTAL);
  float* Vb     = alloc((size_t)128 * P_TOTAL);
  float* aout   = alloc((size_t)128 * P_TOTAL);
  float* hid    = alloc((size_t)64 * P_TOTAL);
  float* wsc    = alloc((size_t)P_TOTAL);
  float* mv     = alloc((size_t)2 * 512);
  float* pooled = alloc((size_t)NB * 512);

  // 0. zero-pad first-layer inputs (xc,xa are adjacent: 64*P floats total)
  {
    const size_t n = (size_t)64 * P_TOTAL;
    zero_fill_kernel<<<(int)((n + 255) / 256), 256, 0, stream>>>(xc, n);
  }

  // 1. geometry + attribute features
  knn_xc_kernel<<<NB * (NPTS / 8), 256, 0, stream>>>(x, xc);
  extract_attrs_kernel<<<(2 * P_TOTAL) / 256, 256, 0, stream>>>(x, xa);

  // 2. coord MLP: 3(pad 32) -> 64 -> 128 -> 256
  run_gemm(cW[0], xc,   cB[0], bufB, 32,  64,  0, stream); run_bn(bufB, mv, cG[0], cBe[0], 64,  stream);
  run_gemm(cW[1], bufB, cB[1], bufA, 64,  128, 0, stream); run_bn(bufA, mv, cG[1], cBe[1], 128, stream);
  run_gemm(cW[2], bufA, cB[2], xall, 128, 256, 0, stream); run_bn(xall, mv, cG[2], cBe[2], 256, stream);

  // 3. attr MLP: 2(pad 32) -> 64 -> 128 -> 256 into xall[256:512]
  float* xall_a = xall + (size_t)256 * P_TOTAL;
  run_gemm(aW[0], xa,   aB[0], bufB,   32,  64,  0, stream); run_bn(bufB,   mv, aG[0], aBe[0], 64,  stream);
  run_gemm(aW[1], bufB, aB[1], bufA,   64,  128, 0, stream); run_bn(bufA,   mv, aG[1], aBe[1], 128, stream);
  run_gemm(aW[2], bufA, aB[2], xall_a, 128, 256, 0, stream); run_bn(xall_a, mv, aG[2], aBe[2], 256, stream);

  // 4. attention
  run_gemm(Wq, xall, bq, Qb, 512, 128, 0, stream);
  run_gemm(Wk, xall, bk, Kb, 512, 128, 0, stream);
  run_gemm(Wv, xall, bv, Vb, 512, 128, 0, stream);
  flash_attn_kernel<<<NB * (NPTS / 128), 256, 0, stream>>>(Qb, Kb, Vb, aout);
  run_gemm(Wfc, aout, bfc, xall, 128, 512, GF_RESID, stream);   // fc + residual

  // 5. attention pooling
  run_gemm(Wp1, xall, bp1, hid, 512, 64, GF_RELU, stream);
  pool_score_kernel<<<(P_TOTAL + 255) / 256, 256, 0, stream>>>(hid, Wp2, bp2, wsc);
  pool_softmax_kernel<<<NB, 256, 0, stream>>>(wsc);
  pool_reduce_kernel<<<NB * 512, 256, 0, stream>>>(xall, wsc, pooled);

  // 6. final fc
  final_fc_kernel<<<(NB * 256 + 255) / 256, 256, 0, stream>>>(pooled, Wf, bf, (float*)d_out);
}
